// MultiHeadAttention_18416819765446
// MI455X (gfx1250) — matrix operand order repaired and checked by reading, not compiled, not run on hardware
//
#include <hip/hip_runtime.h>
#include <hip/hip_bf16.h>

// MI455X / gfx1250: wave32, WMMA 16x16x32 bf16 (fp32 accum).
// Pipeline: prep (weightsT->bf16) -> QKV proj GEMM -> flash attention -> out GEMM.
// Flash attention uses 64-key chunks, exp2-based online softmax (log2e folded
// into the Q projection), and software-pipelined K/V fragment loads.

typedef __bf16 bf16;
typedef __attribute__((ext_vector_type(16))) __bf16 v16bf;
typedef __attribute__((ext_vector_type(8)))  float  v8f;

#define WMMA_BF16(a, b, c) \
  __builtin_amdgcn_wmma_f32_16x16x32_bf16(false, (a), false, (b), (short)0, (c), false, false)

// ---- fragment loaders -------------------------------------------------------
// B fragment (32x16, K x N): lane&15 = N column; lanes 0-15 hold K=0..15,
// lanes 16-31 hold K=16..31, K contiguous across the 16 packed bf16 elements.
static __device__ inline v16bf load16bf(const bf16* p) {
  union { int4 v[2]; v16bf b; } u;
  u.v[0] = *(const int4*)(p);
  u.v[1] = *(const int4*)(p + 8);
  return u.b;
}

// A fragment (16x32, M x K): lane&15 = row M; lanes 0-15: K = k0+0..15,
// lanes 16-31: K = k0+16..31  -> two contiguous 16B chunks per lane.
static __device__ inline v16bf load_a_bf16(const bf16* row, int k0, int half) {
  union { int4 v[2]; v16bf b; } u;
  u.v[0] = *(const int4*)(row + k0 + 16 * half);
  u.v[1] = *(const int4*)(row + k0 + 16 * half + 8);
  return u.b;
}

// Same A pattern but from fp32 memory, converting to bf16 in-register.
static __device__ inline v16bf load_a_f32(const float* row, int k0, int half) {
  float t[16];
  *(float4*)(t + 0)  = *(const float4*)(row + k0 + 16 * half);
  *(float4*)(t + 4)  = *(const float4*)(row + k0 + 16 * half + 4);
  *(float4*)(t + 8)  = *(const float4*)(row + k0 + 16 * half + 8);
  *(float4*)(t + 12) = *(const float4*)(row + k0 + 16 * half + 12);
  v16bf r;
#pragma unroll
  for (int i = 0; i < 16; ++i) r[i] = (bf16)t[i];
  return r;
}

// ---- kernel 0: transpose weight (512x512) to bf16 WT[n][k], fold scale -----
__global__ __launch_bounds__(256) void prep_wt(const float* __restrict__ W,
                                               bf16* __restrict__ WT, float scale) {
  int idx = blockIdx.x * 256 + threadIdx.x;   // 512*512 elements
  int n = idx >> 9, k = idx & 511;
  WT[(size_t)n * 512 + k] = (bf16)(W[(size_t)k * 512 + n] * scale);
}

// ---- kernel 1: projection GEMM: (8192x512 f32) @ WT + bias -> bf16 ---------
// transpose_out==0 : out[((b*8+h)*4096 + t)*64 + d]   (Q, K)
// transpose_out==1 : out[((b*8+h)*64 + d)*4096 + t]   (V transposed for PV)
__global__ __launch_bounds__(128) void proj_gemm(const float* __restrict__ X,
                                                 const bf16* __restrict__ WT,
                                                 const float* __restrict__ bias,
                                                 float bscale,
                                                 bf16* __restrict__ out,
                                                 int transpose_out) {
  const int lane = threadIdx.x & 31;
  const int wave = threadIdx.x >> 5;
  const int tile = blockIdx.x * 4 + wave;  // 0..4095
  const int mtile = tile >> 3;             // 512 row-tiles of 16
  const int ng    = tile & 7;              // 8 col-groups of 64
  const int half  = lane >> 4;
  const int ln    = lane & 15;
  const int row0  = mtile * 16;

  v8f acc[4] = {{}, {}, {}, {}};
  const float* arow = X + (size_t)(row0 + ln) * 512;
  const bf16*  bcol = WT + (size_t)(ng * 64 + ln) * 512 + 16 * half;

  v16bf a = load_a_f32(arow, 0, half);
#pragma unroll 4
  for (int ks = 0; ks < 16; ++ks) {
    v16bf b[4];
#pragma unroll
    for (int nt = 0; nt < 4; ++nt)
      b[nt] = load16bf(bcol + (size_t)nt * 16 * 512 + ks * 32);
    v16bf anext = a;
    if (ks < 15) anext = load_a_f32(arow, (ks + 1) * 32, half);  // prefetch
#pragma unroll
    for (int nt = 0; nt < 4; ++nt) acc[nt] = WMMA_BF16(a, b[nt], acc[nt]);
    a = anext;
  }
#pragma unroll
  for (int nt = 0; nt < 4; ++nt) {
    int ncol = ng * 64 + nt * 16 + ln;
    float bv = bias[ncol] * bscale;
    int h = ncol >> 6, dk = ncol & 63;
#pragma unroll
    for (int r = 0; r < 8; ++r) {
      int row = row0 + r + 8 * half;     // global row in (B*T)
      int b_ = row >> 12, t = row & 4095;
      float val = acc[nt][r] + bv;
      size_t idx = transpose_out
                       ? ((size_t)((b_ * 8 + h) * 64 + dk)) * 4096 + t
                       : ((size_t)((b_ * 8 + h) * 4096 + t)) * 64 + dk;
      out[idx] = (bf16)val;
    }
  }
}

// ---- kernel 2: flash attention. One wave = 16 queries of one (b,h). --------
// 64-key chunks; scores arrive pre-scaled by log2(e)/8 so softmax is pure exp2.
__global__ __launch_bounds__(128) void flash_attn(const bf16* __restrict__ qb,
                                                  const bf16* __restrict__ kb,
                                                  const bf16* __restrict__ vt,
                                                  bf16* __restrict__ ao) {
  __shared__ bf16 lds[4][16 * 64];  // per-wave P tile (16 q x 64 keys)
  const int lane = threadIdx.x & 31;
  const int wave = threadIdx.x >> 5;
  const int w  = blockIdx.x * 4 + wave;  // 0..4095
  const int bh = w >> 8;                 // 256 query-tiles per (b,h)
  const int qt = w & 255;
  const int half = lane >> 4;
  const int ln   = lane & 15;
  const int q0   = qt * 16;

  const bf16* qh = qb + (size_t)bh * 4096 * 64;
  const bf16* kh = kb + (size_t)bh * 4096 * 64;
  const bf16* vh = vt + (size_t)bh * 64 * 4096;
  bf16* pl = lds[wave];

  const bf16* qrow = qh + (size_t)(q0 + ln) * 64;
  v16bf qa0 = load_a_bf16(qrow, 0, half);   // Q pre-scaled by log2(e)/8
  v16bf qa1 = load_a_bf16(qrow, 32, half);

  float m_i[8], l_i[8];
#pragma unroll
  for (int r = 0; r < 8; ++r) { m_i[r] = -3.0e38f; l_i[r] = 0.f; }
  v8f o[4] = {{}, {}, {}, {}};

  for (int kc = 0; kc < 64; ++kc) {  // 4096 keys / 64
    const int k0 = kc * 64;
    // --- S = Q K^T : four 16x16 tiles, software-pipelined K loads ---
    v8f s[4] = {{}, {}, {}, {}};
    {
      const bf16* kr = kh + (size_t)(k0 + ln) * 64 + 16 * half;
      v16bf kb0 = load16bf(kr), kb1 = load16bf(kr + 32);
#pragma unroll
      for (int j = 0; j < 4; ++j) {
        v16bf nb0 = kb0, nb1 = kb1;
        if (j < 3) {
          const bf16* nr = kh + (size_t)(k0 + (j + 1) * 16 + ln) * 64 + 16 * half;
          nb0 = load16bf(nr);
          nb1 = load16bf(nr + 32);
        }
        s[j] = WMMA_BF16(qa0, kb0, s[j]);
        s[j] = WMMA_BF16(qa1, kb1, s[j]);
        kb0 = nb0; kb1 = nb1;
      }
    }
    // --- online softmax (base 2); C-layout: reg r = row (r+8*half), col ln ---
#pragma unroll
    for (int r = 0; r < 8; ++r) {
      float mx = fmaxf(fmaxf(s[0][r], s[1][r]), fmaxf(s[2][r], s[3][r]));
#pragma unroll
      for (int d = 1; d < 16; d <<= 1) mx = fmaxf(mx, __shfl_xor(mx, d, 32));
      float mnew  = fmaxf(m_i[r], mx);
      float alpha = exp2f(m_i[r] - mnew);
      float e0 = exp2f(s[0][r] - mnew);
      float e1 = exp2f(s[1][r] - mnew);
      float e2 = exp2f(s[2][r] - mnew);
      float e3 = exp2f(s[3][r] - mnew);
      float rs = (e0 + e1) + (e2 + e3);
#pragma unroll
      for (int d = 1; d < 16; d <<= 1) rs += __shfl_xor(rs, d, 32);
      l_i[r] = l_i[r] * alpha + rs;
      m_i[r] = mnew;
#pragma unroll
      for (int nt = 0; nt < 4; ++nt) o[nt][r] *= alpha;
      // stage P (bf16) to LDS, row-major 16x64
      bf16* pr = pl + (r + 8 * half) * 64 + ln;
      pr[0]  = (bf16)e0;
      pr[16] = (bf16)e1;
      pr[32] = (bf16)e2;
      pr[48] = (bf16)e3;
    }
    asm volatile("s_wait_dscnt 0x0" ::: "memory");  // P visible to whole wave
    const bf16* prow = pl + (size_t)ln * 64;
    v16bf pa0 = load_a_bf16(prow, 0, half);
    v16bf pa1 = load_a_bf16(prow, 32, half);
    // --- O += P V : V^T layout, pipelined V loads ---
    {
      const bf16* vr = vh + (size_t)ln * 4096 + k0 + 16 * half;
      v16bf vb0 = load16bf(vr), vb1 = load16bf(vr + 32);
#pragma unroll
      for (int nt = 0; nt < 4; ++nt) {
        v16bf nb0 = vb0, nb1 = vb1;
        if (nt < 3) {
          const bf16* nr = vr + (size_t)(nt + 1) * 16 * 4096;
          nb0 = load16bf(nr);
          nb1 = load16bf(nr + 32);
        }
        o[nt] = WMMA_BF16(pa0, vb0, o[nt]);
        o[nt] = WMMA_BF16(pa1, vb1, o[nt]);
        vb0 = nb0; vb1 = nb1;
      }
    }
    asm volatile("s_wait_dscnt 0x0" ::: "memory");  // reads done before rewrite
  }

  // write O/l as bf16 into (B*T, 512) for the output projection
  const int b_ = bh >> 3, h = bh & 7;
#pragma unroll
  for (int nt = 0; nt < 4; ++nt) {
#pragma unroll
    for (int r = 0; r < 8; ++r) {
      int m = r + 8 * half;
      float val = o[nt][r] / l_i[r];
      size_t idx = ((size_t)(b_ * 4096 + q0 + m)) * 512 + h * 64 + nt * 16 + ln;
      ao[idx] = (bf16)val;
    }
  }
}

// ---- kernel 3: output GEMM: (8192x512 bf16) @ WoT + bo -> fp32 d_out -------
__global__ __launch_bounds__(128) void out_gemm(const bf16* __restrict__ A,
                                                const bf16* __restrict__ WT,
                                                const float* __restrict__ bias,
                                                float* __restrict__ out) {
  const int lane = threadIdx.x & 31;
  const int wave = threadIdx.x >> 5;
  const int tile = blockIdx.x * 4 + wave;
  const int mtile = tile >> 3;
  const int ng    = tile & 7;
  const int half  = lane >> 4;
  const int ln    = lane & 15;
  const int row0  = mtile * 16;

  v8f acc[4] = {{}, {}, {}, {}};
  const bf16* arow = A + (size_t)(row0 + ln) * 512;
  const bf16* bcol = WT + (size_t)(ng * 64 + ln) * 512 + 16 * half;

  v16bf a = load_a_bf16(arow, 0, half);
#pragma unroll 4
  for (int ks = 0; ks < 16; ++ks) {
    v16bf b[4];
#pragma unroll
    for (int nt = 0; nt < 4; ++nt)
      b[nt] = load16bf(bcol + (size_t)nt * 16 * 512 + ks * 32);
    v16bf anext = a;
    if (ks < 15) anext = load_a_bf16(arow, (ks + 1) * 32, half);  // prefetch
#pragma unroll
    for (int nt = 0; nt < 4; ++nt) acc[nt] = WMMA_BF16(a, b[nt], acc[nt]);
    a = anext;
  }
#pragma unroll
  for (int nt = 0; nt < 4; ++nt) {
    int ncol = ng * 64 + nt * 16 + ln;
    float bv = bias[ncol];
#pragma unroll
    for (int r = 0; r < 8; ++r) {
      int row = row0 + r + 8 * half;
      out[(size_t)row * 512 + ncol] = acc[nt][r] + bv;
    }
  }
}

extern "C" void kernel_launch(void* const* d_in, const int* in_sizes, int n_in,
                              void* d_out, int out_size, void* d_ws, size_t ws_size,
                              hipStream_t stream) {
  const float* Q  = (const float*)d_in[0];
  const float* K  = (const float*)d_in[1];
  const float* V  = (const float*)d_in[2];
  const float* Wq = (const float*)d_in[3];
  const float* bq = (const float*)d_in[4];
  const float* Wk = (const float*)d_in[5];
  const float* bk = (const float*)d_in[6];
  const float* Wv = (const float*)d_in[7];
  const float* bv = (const float*)d_in[8];
  const float* Wo = (const float*)d_in[9];
  const float* bo = (const float*)d_in[10];
  float* out = (float*)d_out;

  // workspace layout (bf16): 4 transposed weights + q + k + vT + attn_out  (~34 MB)
  bf16* WqT   = (bf16*)d_ws;
  bf16* WkT   = WqT + (size_t)512 * 512;
  bf16* WvT   = WkT + (size_t)512 * 512;
  bf16* WoT   = WvT + (size_t)512 * 512;
  bf16* qbuf  = WoT + (size_t)512 * 512;
  bf16* kbuf  = qbuf  + (size_t)16 * 4096 * 64;   // B*H=16 heads
  bf16* vtbuf = kbuf  + (size_t)16 * 4096 * 64;
  bf16* aobuf = vtbuf + (size_t)16 * 4096 * 64;

  // 0) weights -> bf16 WT[n][k]; fold (1/sqrt(64))*log2(e) into the Q path so
  //    the softmax uses pure exp2 (exact: max/sum share the scaled base).
  const float qscale = 0.125f * 1.44269504088896340736f;
  prep_wt<<<1024, 256, 0, stream>>>(Wq, WqT, qscale);
  prep_wt<<<1024, 256, 0, stream>>>(Wk, WkT, 1.0f);
  prep_wt<<<1024, 256, 0, stream>>>(Wv, WvT, 1.0f);
  prep_wt<<<1024, 256, 0, stream>>>(Wo, WoT, 1.0f);

  // 1) projections (4096 wave-tiles each, 4 waves/block)
  proj_gemm<<<1024, 128, 0, stream>>>(Q, WqT, bq, qscale, qbuf, 0);
  proj_gemm<<<1024, 128, 0, stream>>>(K, WkT, bk, 1.0f,   kbuf, 0);
  proj_gemm<<<1024, 128, 0, stream>>>(V, WvT, bv, 1.0f,   vtbuf, 1);

  // 2) flash attention: 4096 query-tiles (16 q each), 4 waves/block
  flash_attn<<<1024, 128, 0, stream>>>(qbuf, kbuf, vtbuf, aobuf);

  // 3) output projection -> fp32
  out_gemm<<<1024, 128, 0, stream>>>(aobuf, WoT, bo, out);
}
